// SelectiveScan_12240656794149
// MI455X (gfx1250) — matrix operand order repaired and checked by reading, not compiled, not run on hardware
//
#include <hip/hip_runtime.h>

typedef __attribute__((ext_vector_type(2)))  float    v2f;
typedef __attribute__((ext_vector_type(4)))  float    v4f;
typedef __attribute__((ext_vector_type(8)))  float    v8f;
typedef _Float16 half_t;
typedef __attribute__((ext_vector_type(8)))  _Float16 v8h;
typedef __attribute__((ext_vector_type(16))) _Float16 v16h;

#define D_MODEL 768
#define D_STATE 16
#define SEQ_L   4096
#define BATCH   16
#define M_ROWS  (BATCH * SEQ_L)   // 65536

union AFrag { v16h v; v8h h[2]; };

__device__ __forceinline__ v8f vzero8() {
  v8f z;
#pragma unroll
  for (int i = 0; i < 8; i++) z[i] = 0.0f;
  return z;
}

// fp32 WMMA: D = A(16x4) x B(4x16) + C
__device__ __forceinline__ v8f wmma4(v2f a, v2f b, v8f c) {
  return __builtin_amdgcn_wmma_f32_16x16x4_f32(false, a, false, b, (short)0, c,
                                               false, false);
}
// f16 WMMA: D = A(16x32 f16) x B(32x16 f16) + C(f32)
__device__ __forceinline__ v8f wmma16(v16h a, v16h b, v8f c) {
  return __builtin_amdgcn_wmma_f32_16x16x32_f16(false, a, false, b, (short)0, c,
                                                false, false);
}

// ---------------------------------------------------------------------------
// proj_in with split-f16 WMMA:
// U[m,e] = sum_d X[m,d]*W[e,d] + bias[e]
// A (X) is split fp32->f16 hi/lo on the fly (VALU co-executes with WMMA);
// B comes from precomputed Whi/Wlo planes. Epilogue writes Uhi/Ulo planes.
// block 256 = 8 waves (4M x 2N), wave tile 32x64, block tile 128x128.
// ---------------------------------------------------------------------------
__global__ __launch_bounds__(256) void k_gemm_in(
    const float* __restrict__ X,
    const half_t* __restrict__ Whi, const half_t* __restrict__ Wlo,
    const float* __restrict__ bias,
    half_t* __restrict__ Uhi, half_t* __restrict__ Ulo) {
  const int lane  = threadIdx.x & 31;
  const int wave  = threadIdx.x >> 5;
  const int lm    = lane & 15;
  const int lhalf = lane >> 4;
  const int mbase = blockIdx.y * 128 + (wave & 3) * 32;
  const int nbase = blockIdx.x * 128 + (wave >> 2) * 64;

  const float* xr[2];
#pragma unroll
  for (int mt = 0; mt < 2; mt++)
    xr[mt] = X + (mbase + mt * 16 + lm) * D_MODEL + 16 * lhalf;
  const half_t* wrh[4];
  const half_t* wrl[4];
#pragma unroll
  for (int nt = 0; nt < 4; nt++) {
    const int n = nbase + nt * 16 + lm;
    wrh[nt] = Whi + n * D_MODEL + 16 * lhalf;
    wrl[nt] = Wlo + n * D_MODEL + 16 * lhalf;
  }

  v8f c[2][4];
#pragma unroll
  for (int i = 0; i < 2; i++)
#pragma unroll
    for (int j = 0; j < 4; j++) c[i][j] = vzero8();

  for (int kb = 0; kb < D_MODEL; kb += 32) {
    v16h bh[4], bl[4];
#pragma unroll
    for (int nt = 0; nt < 4; nt++) {
      bh[nt] = *(const v16h*)(wrh[nt] + kb);
      bl[nt] = *(const v16h*)(wrl[nt] + kb);
    }
    v16h ah[2], al[2];
#pragma unroll
    for (int mt = 0; mt < 2; mt++) {
      const float* p = xr[mt] + kb;
      float f[16];
      *(v4f*)(f + 0)  = *(const v4f*)(p + 0);
      *(v4f*)(f + 4)  = *(const v4f*)(p + 4);
      *(v4f*)(f + 8)  = *(const v4f*)(p + 8);
      *(v4f*)(f + 12) = *(const v4f*)(p + 12);
      v16h hi, lo;
#pragma unroll
      for (int i = 0; i < 16; i++) {
        half_t h = (half_t)f[i];
        hi[i] = h;
        lo[i] = (half_t)(f[i] - (float)h);
      }
      ah[mt] = hi; al[mt] = lo;
    }
#pragma unroll
    for (int mt = 0; mt < 2; mt++)
#pragma unroll
      for (int nt = 0; nt < 4; nt++) {
        v8f acc = c[mt][nt];
        acc = wmma16(al[mt], bh[nt], acc);   // lo*hi
        acc = wmma16(ah[mt], bl[nt], acc);   // hi*lo
        acc = wmma16(ah[mt], bh[nt], acc);   // hi*hi
        c[mt][nt] = acc;
      }
  }

#pragma unroll
  for (int mt = 0; mt < 2; mt++)
#pragma unroll
    for (int nt = 0; nt < 4; nt++) {
      const int ncol = nbase + nt * 16 + lm;
      const float bn = bias[ncol];
      const int row0 = mbase + mt * 16 + 8 * lhalf;
#pragma unroll
      for (int r = 0; r < 8; r++) {
        float v = c[mt][nt][r] + bn;
        half_t h = (half_t)v;
        const int idx = (row0 + r) * D_MODEL + ncol;
        Uhi[idx] = h;
        Ulo[idx] = (half_t)(v - (float)h);
      }
    }
}

// ---------------------------------------------------------------------------
// proj_out with split-f16 WMMA + fp32 WMMA tail:
// OUT[m,e] = sum_d U[m,d]*W2[e,d] + sum_n HS[m,n]*G[n,e] + bout[e]
// ---------------------------------------------------------------------------
__global__ __launch_bounds__(256) void k_gemm_out(
    const half_t* __restrict__ Uhi, const half_t* __restrict__ Ulo,
    const half_t* __restrict__ W2hi, const half_t* __restrict__ W2lo,
    const float* __restrict__ HS, const float* __restrict__ G,
    const float* __restrict__ bout, float* __restrict__ OUT) {
  const int lane  = threadIdx.x & 31;
  const int wave  = threadIdx.x >> 5;
  const int lm    = lane & 15;
  const int lhalf = lane >> 4;
  const int lk    = lhalf << 1;
  const int mbase = blockIdx.y * 128 + (wave & 3) * 32;
  const int nbase = blockIdx.x * 128 + (wave >> 2) * 64;

  const half_t* urh[2];
  const half_t* url[2];
#pragma unroll
  for (int mt = 0; mt < 2; mt++) {
    const int m = mbase + mt * 16 + lm;
    urh[mt] = Uhi + m * D_MODEL + 16 * lhalf;
    url[mt] = Ulo + m * D_MODEL + 16 * lhalf;
  }
  const half_t* wrh[4];
  const half_t* wrl[4];
#pragma unroll
  for (int nt = 0; nt < 4; nt++) {
    const int n = nbase + nt * 16 + lm;
    wrh[nt] = W2hi + n * D_MODEL + 16 * lhalf;
    wrl[nt] = W2lo + n * D_MODEL + 16 * lhalf;
  }

  v8f c[2][4];
#pragma unroll
  for (int i = 0; i < 2; i++)
#pragma unroll
    for (int j = 0; j < 4; j++) c[i][j] = vzero8();

  for (int kb = 0; kb < D_MODEL; kb += 32) {
    v16h bh[4], bl[4];
#pragma unroll
    for (int nt = 0; nt < 4; nt++) {
      bh[nt] = *(const v16h*)(wrh[nt] + kb);
      bl[nt] = *(const v16h*)(wrl[nt] + kb);
    }
    v16h ah[2], al[2];
#pragma unroll
    for (int mt = 0; mt < 2; mt++) {
      AFrag fh, fl;
      fh.h[0] = *(const v8h*)(urh[mt] + kb);
      fh.h[1] = *(const v8h*)(urh[mt] + kb + 8);
      fl.h[0] = *(const v8h*)(url[mt] + kb);
      fl.h[1] = *(const v8h*)(url[mt] + kb + 8);
      ah[mt] = fh.v; al[mt] = fl.v;
    }
#pragma unroll
    for (int mt = 0; mt < 2; mt++)
#pragma unroll
      for (int nt = 0; nt < 4; nt++) {
        v8f acc = c[mt][nt];
        acc = wmma16(al[mt], bh[nt], acc);
        acc = wmma16(ah[mt], bl[nt], acc);
        acc = wmma16(ah[mt], bh[nt], acc);
        c[mt][nt] = acc;
      }
  }

  // K=16 tail in full fp32: + HS @ G
  const float* hr0 = HS + (mbase + lm) * D_STATE + lk;
  const float* hr1 = hr0 + 16 * D_STATE;
#pragma unroll
  for (int kb = 0; kb < D_STATE; kb += 4) {
    v2f a0 = *(const v2f*)(hr0 + kb);
    v2f a1 = *(const v2f*)(hr1 + kb);
#pragma unroll
    for (int nt = 0; nt < 4; nt++) {
      const int ncol = nbase + nt * 16 + lm;
      v2f b;
      b.x = G[(kb + lk) * D_MODEL + ncol];
      b.y = G[(kb + lk + 1) * D_MODEL + ncol];
      c[0][nt] = wmma4(a0, b, c[0][nt]);
      c[1][nt] = wmma4(a1, b, c[1][nt]);
    }
  }

#pragma unroll
  for (int mt = 0; mt < 2; mt++)
#pragma unroll
    for (int nt = 0; nt < 4; nt++) {
      const int ncol = nbase + nt * 16 + lm;
      const float bn = bout[ncol];
      float* o = OUT + (mbase + mt * 16 + 8 * lhalf) * D_MODEL + ncol;
#pragma unroll
      for (int r = 0; r < 8; r++) o[r * D_MODEL] = c[mt][nt][r] + bn;
    }
}

// ---------------------------------------------------------------------------
// Bu[m,n] = sum_d X[m,d]*M2[n,d] + bu0[n]  -- full fp32 (feeds the recurrence)
// ---------------------------------------------------------------------------
__global__ __launch_bounds__(256) void k_bu(
    const float* __restrict__ X, const float* __restrict__ M2,
    const float* __restrict__ bu0, float* __restrict__ Bu) {
  const int lane  = threadIdx.x & 31;
  const int wave  = threadIdx.x >> 5;
  const int lm    = lane & 15;
  const int lhalf = lane >> 4;
  const int lk    = lhalf << 1;
  const int mbase = blockIdx.x * 512 + wave * 64;

  const float* xr[4];
#pragma unroll
  for (int mt = 0; mt < 4; mt++)
    xr[mt] = X + (mbase + mt * 16 + lm) * D_MODEL + lk;
  const float* br = M2 + lm * D_MODEL + lk;

  v8f c[4];
#pragma unroll
  for (int i = 0; i < 4; i++) c[i] = vzero8();

  for (int kb = 0; kb < D_MODEL; kb += 4) {
    v2f b = *(const v2f*)(br + kb);
    v2f a0 = *(const v2f*)(xr[0] + kb);
    v2f a1 = *(const v2f*)(xr[1] + kb);
    v2f a2 = *(const v2f*)(xr[2] + kb);
    v2f a3 = *(const v2f*)(xr[3] + kb);
    c[0] = wmma4(a0, b, c[0]);
    c[1] = wmma4(a1, b, c[1]);
    c[2] = wmma4(a2, b, c[2]);
    c[3] = wmma4(a3, b, c[3]);
  }

  const float bn = bu0[lm];
#pragma unroll
  for (int mt = 0; mt < 4; mt++) {
    float* o = Bu + (mbase + mt * 16 + 8 * lhalf) * D_STATE + lm;
#pragma unroll
    for (int r = 0; r < 8; r++) o[r * D_STATE] = c[mt][r] + bn;
  }
}

// ---------------------------------------------------------------------------
// Serial recurrence: h_t = silu(h_{t-1} @ A + Bu_t). One wave per batch,
// lane n owns h[n]; Bu software-pipelined 8 steps ahead.
// ---------------------------------------------------------------------------
__global__ __launch_bounds__(32) void k_scan(
    const float* __restrict__ Bu, const float* __restrict__ A,
    float* __restrict__ HS) {
  const int b = blockIdx.x;
  const int lane = threadIdx.x & 31;
  const int n = lane & 15;

  float acol[16];
#pragma unroll
  for (int k = 0; k < 16; k++) acol[k] = A[k * 16 + n];

  const float* bu = Bu + b * SEQ_L * D_STATE;
  float* hs = HS + b * SEQ_L * D_STATE;

  float h = 0.0f;
  float nxt[8];
#pragma unroll
  for (int t = 0; t < 8; t++) nxt[t] = bu[t * D_STATE + n];

  for (int cchunk = 0; cchunk < SEQ_L / 8; cchunk++) {
    float cur[8];
#pragma unroll
    for (int t = 0; t < 8; t++) cur[t] = nxt[t];
    if (cchunk + 1 < SEQ_L / 8) {
      const float* p = bu + (cchunk + 1) * 8 * D_STATE;
#pragma unroll
      for (int t = 0; t < 8; t++) nxt[t] = p[t * D_STATE + n];
    }
#pragma unroll
    for (int t = 0; t < 8; t++) {
      float acc0 = cur[t], acc1 = 0.f, acc2 = 0.f, acc3 = 0.f;
#pragma unroll
      for (int k = 0; k < 16; k += 4) {
        acc0 = fmaf(__shfl(h, k + 0, 32), acol[k + 0], acc0);
        acc1 = fmaf(__shfl(h, k + 1, 32), acol[k + 1], acc1);
        acc2 = fmaf(__shfl(h, k + 2, 32), acol[k + 2], acc2);
        acc3 = fmaf(__shfl(h, k + 3, 32), acol[k + 3], acc3);
      }
      float v = (acc0 + acc1) + (acc2 + acc3);
      float ex = __builtin_amdgcn_exp2f(v * -1.4426950408889634f);
      h = v * __builtin_amdgcn_rcpf(1.0f + ex);
      if (lane < 16) hs[(cchunk * 8 + t) * D_STATE + n] = h;
    }
  }
}

// ---------------------------------------------------------------------------
// Prep kernels
// ---------------------------------------------------------------------------
__global__ void k_split_w(const float* __restrict__ Win,
                          half_t* __restrict__ Whi, half_t* __restrict__ Wlo) {
  int i = blockIdx.x * 256 + threadIdx.x;
  if (i < D_MODEL * D_MODEL) {
    float v = Win[i];
    half_t h = (half_t)v;
    Whi[i] = h;
    Wlo[i] = (half_t)(v - (float)h);
  }
}

__global__ void k_w2(const float* __restrict__ Wout, const float* __restrict__ Dv,
                     half_t* __restrict__ W2hi, half_t* __restrict__ W2lo) {
  int i = blockIdx.x * 256 + threadIdx.x;
  if (i < D_MODEL * D_MODEL) {
    float v = Wout[i] * Dv[i % D_MODEL];
    half_t h = (half_t)v;
    W2hi[i] = h;
    W2lo[i] = (half_t)(v - (float)h);
  }
}

__global__ void k_g(const float* __restrict__ Wout, const float* __restrict__ C,
                    float* __restrict__ G) {
  int i = blockIdx.x * 256 + threadIdx.x;
  if (i >= D_STATE * D_MODEL) return;
  int n = i & 15, e = i >> 4;
  float s = 0.f;
  for (int d = 0; d < D_MODEL; d++)
    s = fmaf(Wout[e * D_MODEL + d], C[d * D_STATE + n], s);
  G[n * D_MODEL + e] = s;
}

__global__ void k_m2(const float* __restrict__ Bm, const float* __restrict__ Win,
                     const float* __restrict__ bin, float* __restrict__ M2,
                     float* __restrict__ bu0) {
  int i = blockIdx.x * 256 + threadIdx.x;
  if (i < D_STATE * D_MODEL) {
    int nn = i / D_MODEL, dd = i % D_MODEL;
    float s = 0.f;
    for (int e = 0; e < D_MODEL; e++)
      s = fmaf(Bm[nn * D_MODEL + e], Win[e * D_MODEL + dd], s);
    M2[nn * D_MODEL + dd] = s;
  }
  if (i < D_STATE) {
    float s = 0.f;
    for (int e = 0; e < D_MODEL; e++) s = fmaf(bin[e], Bm[i * D_MODEL + e], s);
    bu0[i] = s;
  }
}

// ---------------------------------------------------------------------------
extern "C" void kernel_launch(void* const* d_in, const int* in_sizes, int n_in,
                              void* d_out, int out_size, void* d_ws, size_t ws_size,
                              hipStream_t stream) {
  (void)in_sizes; (void)n_in; (void)out_size; (void)ws_size;
  const float* x    = (const float*)d_in[0];
  const float* Win  = (const float*)d_in[1];
  const float* bin  = (const float*)d_in[2];
  const float* A    = (const float*)d_in[3];
  const float* Bm   = (const float*)d_in[4];
  const float* C    = (const float*)d_in[5];
  const float* Dv   = (const float*)d_in[6];
  const float* Wout = (const float*)d_in[7];
  const float* bout = (const float*)d_in[8];
  float* OUT = (float*)d_out;

  char* p = (char*)d_ws;
  half_t* Uhi = (half_t*)p;  p += (size_t)M_ROWS * D_MODEL * 2;   // 100.7 MB
  half_t* Ulo = (half_t*)p;  p += (size_t)M_ROWS * D_MODEL * 2;   // 100.7 MB
  float*  Bu  = (float*)p;   p += (size_t)M_ROWS * D_STATE * 4;   // 4 MB
  float*  HS  = (float*)p;   p += (size_t)M_ROWS * D_STATE * 4;   // 4 MB
  half_t* Whi = (half_t*)p;  p += (size_t)D_MODEL * D_MODEL * 2;
  half_t* Wlo = (half_t*)p;  p += (size_t)D_MODEL * D_MODEL * 2;
  half_t* W2hi= (half_t*)p;  p += (size_t)D_MODEL * D_MODEL * 2;
  half_t* W2lo= (half_t*)p;  p += (size_t)D_MODEL * D_MODEL * 2;
  float*  G   = (float*)p;   p += (size_t)D_STATE * D_MODEL * 4;
  float*  M2  = (float*)p;   p += (size_t)D_STATE * D_MODEL * 4;
  float*  bu0 = (float*)p;

  const int wele = D_MODEL * D_MODEL;
  k_split_w<<<(wele + 255) / 256, 256, 0, stream>>>(Win, Whi, Wlo);
  k_w2<<<(wele + 255) / 256, 256, 0, stream>>>(Wout, Dv, W2hi, W2lo);
  k_g <<<(D_STATE * D_MODEL + 255) / 256, 256, 0, stream>>>(Wout, C, G);
  k_m2<<<(D_STATE * D_MODEL + 255) / 256, 256, 0, stream>>>(Bm, Win, bin, M2, bu0);

  // U(hi/lo) = split( x @ W_in^T + b_in )
  k_gemm_in<<<dim3(D_MODEL / 128, M_ROWS / 128), 256, 0, stream>>>(
      x, Whi, Wlo, bin, Uhi, Ulo);
  // Bu = x @ M2^T + bu0   (fp32)
  k_bu<<<M_ROWS / 512, 256, 0, stream>>>(x, M2, bu0, Bu);
  // serial scan -> HS
  k_scan<<<BATCH, 32, 0, stream>>>(Bu, A, HS);
  // OUT = U @ W2^T + HS @ G + b_out
  k_gemm_out<<<dim3(D_MODEL / 128, M_ROWS / 128), 256, 0, stream>>>(
      Uhi, Ulo, W2hi, W2lo, HS, G, bout, OUT);
}
